// RNN_FunctionValue_20469814132833
// MI455X (gfx1250) — matrix operand order repaired and checked by reading, not compiled, not run on hardware
//
#include <hip/hip_runtime.h>
#include <math.h>

// ---------------------------------------------------------------------------
// Persistent bf16-WMMA RNN for MI455X (gfx1250, wave32).
//   h_{t+1} = tanh( x_t*W_ih + b_ih + b_hh + h_t @ W_hh^T )
// B=64, T=1024, H=512. 16 workgroups, each owns a 32-column slice of W_hh^T
// pinned in LDS (32KB bf16). h ping-pongs in workspace (L2 resident).
// Device-wide barrier per timestep (16 resident WGs -> spin barrier is safe).
// Each wave computes one 16x16 tile via 16x v_wmma_f32_16x16x32_bf16 per step.
// ---------------------------------------------------------------------------

typedef __bf16 bf16_t;
typedef __attribute__((ext_vector_type(8)))  bf16_t v8bf;
typedef __attribute__((ext_vector_type(16))) bf16_t v16bf;
typedef __attribute__((ext_vector_type(8)))  float  v8f;

#define BB   64
#define TT   1024
#define HH   512
#define NWG  16
#define NCOL (HH / NWG)            // 32 output columns per workgroup

// workspace layout (bytes)
#define CNT_OFF    0               // u32 barrier counter, u32 generation
#define WPACK_OFF  256             // bf16 [HH][HH]  (B matrix: Wpack[n][k] = W_hh[n][k])
#define WPACK_BYTES (HH * HH * 2)  // 524288
#define HBUF_OFF   (WPACK_OFF + WPACK_BYTES)
#define HBUF_ELEMS (BB * HH)       // one buffer, bf16
// two buffers -> 2*64*512*2 = 131072 bytes; total ws use ~656 KB

// ---------------------------------------------------------------------------
__global__ void rnn_init_prepack(const float* __restrict__ W_hh,
                                 unsigned* __restrict__ barrier_ws,
                                 bf16_t* __restrict__ wpack,
                                 bf16_t* __restrict__ hbuf) {
    const int tid = blockIdx.x * blockDim.x + threadIdx.x;   // 65536 threads
    // convert W_hh (row-major [n][k]) to bf16, identity layout
#pragma unroll
    for (int r = 0; r < 4; ++r) {
        int idx = tid + r * 65536;
        if (idx < HH * HH) wpack[idx] = (bf16_t)W_hh[idx];
    }
    // zero both h ping-pong buffers (h0 == 0 per reference)
#pragma unroll
    for (int r = 0; r < 2; ++r) {
        int idx = tid + r * 65536;
        if (idx < 2 * HBUF_ELEMS) hbuf[idx] = (bf16_t)0.0f;
    }
    if (tid == 0) { barrier_ws[0] = 0u; barrier_ws[1] = 0u; }
}

// ---------------------------------------------------------------------------
__device__ __forceinline__ void grid_sync(unsigned* cnt, volatile unsigned* gen) {
    __threadfence();          // make this WG's h stores visible device-wide
    __syncthreads();
    if (threadIdx.x == 0) {
        unsigned g = *gen;
        if (atomicAdd(cnt, 1u) == NWG - 1) {
            *cnt = 0u;
            __threadfence();
            *gen = g + 1u;    // release all spinners
        } else {
            while (*gen == g) { __builtin_amdgcn_s_sleep(1); }
        }
    }
    __syncthreads();
}

// ---------------------------------------------------------------------------
__global__ void __launch_bounds__(256)
rnn_recurrence(const float* __restrict__ inputs,   // [B][T]
               const float* __restrict__ W_ih,     // [H]
               const float* __restrict__ b_ih,     // [H]
               const float* __restrict__ b_hh,     // [H]
               const bf16_t* __restrict__ wpack,   // [H][H] bf16
               bf16_t* __restrict__ hbuf,          // [2][B][H] bf16 ping-pong
               float* __restrict__ state_out,      // [B][H] fp32 (d_out + 64)
               unsigned* __restrict__ barrier_ws) {
    __shared__ bf16_t lds_w[NCOL * HH];            // 32 KB: this WG's W slice

    const int wg   = blockIdx.x;                   // 0..15 -> H-slice
    const int lane = threadIdx.x & 31;
    const int wave = threadIdx.x >> 5;             // 0..7
    const int half = lane >> 4;                    // 0/1: lane half (ISA layout)
    const int l15  = lane & 15;

    // stage W slice into LDS once (coalesced v8bf -> ds_store_b128)
    {
        const v8bf* src = (const v8bf*)(wpack + (size_t)wg * NCOL * HH);
        v8bf* dst = (v8bf*)lds_w;
        for (int i = threadIdx.x; i < (NCOL * HH) / 8; i += 256) dst[i] = src[i];
    }
    __syncthreads();

    // tile assignment: 4 M-tiles x 2 N-tiles, one per wave
    const int m0       = (wave & 3) * 16;          // batch-row base
    const int ntile    = wave >> 2;                // 0/1
    const int n_local0 = ntile * 16;
    const int n_global = wg * NCOL + n_local0 + l15;   // this lane's N (C/D layout)
    const int n_lds    = n_local0 + l15;

    // loop-invariant per-lane epilogue constants
    const float wih_n  = W_ih[n_global];
    const float bias_n = b_ih[n_global] + b_hh[n_global];

    unsigned* cnt = barrier_ws;
    volatile unsigned* gen = barrier_ws + 1;

    for (int t = 0; t < TT; ++t) {
        const bf16_t* hsrc = hbuf + (size_t)(t & 1) * HBUF_ELEMS;
        bf16_t*       hdst = hbuf + (size_t)((t + 1) & 1) * HBUF_ELEMS;

        if (t + 16 < TT)  // warm next input samples for this wave's rows
            __builtin_prefetch(&inputs[(m0 + half * 8) * TT + t + 16], 0, 0);

        v8f acc = {};
#pragma unroll
        for (int k0 = 0; k0 < HH; k0 += 32) {
            // A tile (h, 16x32 bf16): lanes 0-15 -> M=l15, K=k0+{0..15};
            //                          lanes 16-31 -> K=k0+{16..31}
            const v8bf* pa = (const v8bf*)(hsrc + (size_t)(m0 + l15) * HH + k0 + half * 16);
            v8bf a_lo = pa[0];
            v8bf a_hi = pa[1];                      // +8 elements
            v16bf a = __builtin_shufflevector(a_lo, a_hi,
                0,1,2,3,4,5,6,7,8,9,10,11,12,13,14,15);

            // B tile (W^T, 32x16 bf16): lane N=l15; lanes 0-15 K=k0..k0+15,
            //                           lanes 16-31 K=k0+16..k0+31 (contiguous)
            const v8bf* pb = (const v8bf*)(lds_w + (size_t)n_lds * HH + k0 + half * 16);
            v8bf b_lo = pb[0];
            v8bf b_hi = pb[1];
            v16bf b = __builtin_shufflevector(b_lo, b_hi,
                0,1,2,3,4,5,6,7,8,9,10,11,12,13,14,15);

            acc = __builtin_amdgcn_wmma_f32_16x16x32_bf16(
                false, a, false, b, (short)0, acc, false, false);
        }

        // epilogue: C/D layout -> VGPR i holds M = m0 + i + 8*half, N = n_global
#pragma unroll
        for (int i = 0; i < 8; ++i) {
            const int m = m0 + i + half * 8;
            const float x = inputs[(size_t)m * TT + t];
            float v = tanhf(acc[i] + x * wih_n + bias_n);
            hdst[(size_t)m * HH + n_global] = (bf16_t)v;
            if (t == TT - 1) state_out[(size_t)m * HH + n_global] = v;
        }

        grid_sync(cnt, gen);
    }
}

// ---------------------------------------------------------------------------
__global__ void rnn_out_proj(const float* __restrict__ state,  // [B][H]
                             const float* __restrict__ W_out,  // [H]
                             const float* __restrict__ b_out,  // [1]
                             float* __restrict__ out) {        // [B]
    __shared__ float red[64];
    const int b = blockIdx.x;
    float s = 0.0f;
    for (int n = threadIdx.x; n < HH; n += 64)
        s += state[(size_t)b * HH + n] * W_out[n];
    red[threadIdx.x] = s;
    __syncthreads();
    for (int off = 32; off > 0; off >>= 1) {
        if (threadIdx.x < off) red[threadIdx.x] += red[threadIdx.x + off];
        __syncthreads();
    }
    if (threadIdx.x == 0) out[b] = red[0] + b_out[0];
}

// ---------------------------------------------------------------------------
extern "C" void kernel_launch(void* const* d_in, const int* in_sizes, int n_in,
                              void* d_out, int out_size, void* d_ws, size_t ws_size,
                              hipStream_t stream) {
    const float* inputs = (const float*)d_in[0];   // [B,T,1]
    // d_in[1] = state (ignored by reference forward)
    const float* W_ih   = (const float*)d_in[2];   // [H,1]
    const float* W_hh   = (const float*)d_in[3];   // [H,H]
    const float* b_ih   = (const float*)d_in[4];   // [H]
    const float* b_hh   = (const float*)d_in[5];   // [H]
    const float* W_out  = (const float*)d_in[6];   // [1,H]
    const float* b_out  = (const float*)d_in[7];   // [1]

    float* out       = (float*)d_out;              // [B] outputs
    float* state_out = (float*)d_out + BB;         // [B,H] new_state

    char* ws = (char*)d_ws;
    unsigned* barrier_ws = (unsigned*)(ws + CNT_OFF);
    bf16_t*   wpack      = (bf16_t*)(ws + WPACK_OFF);
    bf16_t*   hbuf       = (bf16_t*)(ws + HBUF_OFF);

    rnn_init_prepack<<<256, 256, 0, stream>>>(W_hh, barrier_ws, wpack, hbuf);
    rnn_recurrence<<<NWG, 256, 0, stream>>>(inputs, W_ih, b_ih, b_hh,
                                            wpack, hbuf, state_out, barrier_ws);
    rnn_out_proj<<<BB, 64, 0, stream>>>(state_out, W_out, b_out, out);
}
